// NeuronTransformer_489626271831
// MI455X (gfx1250) — compile-verified, matrix operand order checked by reading, not run on hardware
//
#include <hip/hip_runtime.h>
#include <hip/hip_bf16.h>

typedef __attribute__((ext_vector_type(16))) _Float16 v16h;
typedef __attribute__((ext_vector_type(8)))  float    v8f;

#define B_   2
#define T_   1024
#define C_   1024
#define H_   16
#define L_   8
#define V_   4096
#define D_   64
#define EPS_ 1e-5f

// ---------------- embedding: x = wte[tok] + wpe ----------------
__global__ void k_embed(const int* __restrict__ tok, const float* __restrict__ wte,
                        const float* __restrict__ wpe, float* __restrict__ x) {
  int row = blockIdx.x;                 // b*T + t
  int t   = row % T_;
  int tk  = tok[row];
  const float* src = wte + (size_t)tk * C_;
  const float* pos = wpe + (size_t)t  * C_;
  float* dst = x + (size_t)row * C_;
  for (int i = threadIdx.x; i < C_; i += blockDim.x)
    dst[i] = src[i] + pos[i];
}

// ---------------- layernorm: fp32 in -> f16 out ----------------
__global__ void k_layernorm(const float* __restrict__ x, const float* __restrict__ g,
                            const float* __restrict__ b, _Float16* __restrict__ out) {
  __shared__ float red[256];
  int row = blockIdx.x;
  int tid = threadIdx.x;
  const float* xr = x + (size_t)row * C_;
  float v0[4];
  float s = 0.f;
  for (int i = 0; i < 4; ++i) { v0[i] = xr[tid + i * 256]; s += v0[i]; }
  red[tid] = s; __syncthreads();
  for (int o = 128; o > 0; o >>= 1) { if (tid < o) red[tid] += red[tid + o]; __syncthreads(); }
  float mean = red[0] * (1.f / C_);
  __syncthreads();
  s = 0.f;
  for (int i = 0; i < 4; ++i) { float d = v0[i] - mean; s += d * d; }
  red[tid] = s; __syncthreads();
  for (int o = 128; o > 0; o >>= 1) { if (tid < o) red[tid] += red[tid + o]; __syncthreads(); }
  float rstd = rsqrtf(red[0] * (1.f / C_) + EPS_);
  _Float16* orow = out + (size_t)row * C_;
  for (int i = 0; i < 4; ++i) {
    int c = tid + i * 256;
    orow[c] = (_Float16)((v0[i] - mean) * rstd * g[c] + b[c]);
  }
}

// ------- weight convert + transpose: f32[K,N] -> f16[N,K] -------
__global__ void k_convT(const float* __restrict__ w, _Float16* __restrict__ wt,
                        int Kd, int Nd) {
  size_t idx = (size_t)blockIdx.x * blockDim.x + threadIdx.x;
  size_t tot = (size_t)Kd * Nd;
  if (idx >= tot) return;
  int n = (int)(idx % Nd);
  int k = (int)(idx / Nd);
  wt[(size_t)n * Kd + k] = (_Float16)w[idx];
}

// ------- split qkv fp32 [B,T,3C] -> Q,K head-major f16, V transposed f16 -------
// VT is padded to 128 rows per (b,h): rows 64..127 are never read into a stored
// output column (epilogue guards n < Nd), so no zero-fill is required.
__global__ void k_split_qkv(const float* __restrict__ qkv, _Float16* __restrict__ q,
                            _Float16* __restrict__ k, _Float16* __restrict__ vt) {
  size_t idx = (size_t)blockIdx.x * blockDim.x + threadIdx.x;   // over B*T*C
  int c = (int)(idx % C_);
  size_t bt = idx / C_;
  int t = (int)(bt % T_);
  int b = (int)(bt / T_);
  int h = c / D_, d = c % D_;
  const float* src = qkv + bt * (size_t)(3 * C_);
  size_t bh = (size_t)b * H_ + h;
  q [(bh * T_ + t) * D_ + d] = (_Float16)src[c];
  k [(bh * T_ + t) * D_ + d] = (_Float16)src[C_ + c];
  vt[(bh * 128 + d) * T_ + t] = (_Float16)src[2 * C_ + c];
}

// ------- causal softmax, in-place on f16 score rows -------
__global__ void k_softmax_causal(_Float16* __restrict__ p, float scale) {
  __shared__ float red[256];
  int row  = blockIdx.x;               // bh*T + qpos
  int qpos = row % T_;
  int tid  = threadIdx.x;
  _Float16* pr = p + (size_t)row * T_;
  float v[4];
  float mx = -3.0e38f;
  for (int i = 0; i < 4; ++i) {
    int kk = tid + i * 256;
    float s = (kk <= qpos) ? (float)pr[kk] * scale : -3.0e38f;
    v[i] = s; mx = fmaxf(mx, s);
  }
  red[tid] = mx; __syncthreads();
  for (int o = 128; o > 0; o >>= 1) { if (tid < o) red[tid] = fmaxf(red[tid], red[tid + o]); __syncthreads(); }
  mx = red[0]; __syncthreads();
  float s = 0.f;
  for (int i = 0; i < 4; ++i) {
    int kk = tid + i * 256;
    float e = (kk <= qpos) ? __expf(v[i] - mx) : 0.f;
    v[i] = e; s += e;
  }
  red[tid] = s; __syncthreads();
  for (int o = 128; o > 0; o >>= 1) { if (tid < o) red[tid] += red[tid + o]; __syncthreads(); }
  float inv = 1.f / red[0];
  for (int i = 0; i < 4; ++i) pr[tid + i * 256] = (_Float16)(v[i] * inv);
}

// ------- merge heads: y fp32 [B,H,T,D] -> f16 [B,T,C] -------
__global__ void k_merge_heads(const float* __restrict__ y, _Float16* __restrict__ out) {
  size_t idx = (size_t)blockIdx.x * blockDim.x + threadIdx.x;   // over B*T*C
  int c = (int)(idx % C_);
  size_t bt = idx / C_;
  int t = (int)(bt % T_);
  int b = (int)(bt / T_);
  int h = c / D_, d = c % D_;
  out[idx] = (_Float16)y[(((size_t)b * H_ + h) * T_ + t) * D_ + d];
}

// ---------------- WMMA GEMM with async global->LDS double buffering ----------
// C[m,n] = sum_k A[m,k] * Bt[n,k]  (+bias[n]) (opt. tanh-GELU)
// A: f16 [Md,Kd] row-major (Md multiple of 128). Bt: f16 [>=ceil128(Nd),Kd].
// Batched via blockIdx.z with element strides sA,sB,sO.
// Block 256 thr = 8 waves; tile 128x128; BK=32; wave sub-tile 32x64.
union F16Frag { uint4 u[2]; v16h v; };

__device__ __forceinline__ v16h lds_frag(const _Float16* s, int rowBase, int lr, int kg) {
  // 16-bit A/B fragment (ISA 7.12.2): lane<16 holds K 0..7 & 16..23,
  // lane>=16 holds K 8..15 & 24..31 for row (lane&15). Two aligned 16B LDS reads.
  int base = (rowBase + lr) * 48 + kg * 8;
  F16Frag f;
  f.u[0] = *(const uint4*)(s + base);
  f.u[1] = *(const uint4*)(s + base + 16);
  return f.v;
}

// one 16-byte async copy: global -> LDS, tracked by ASYNCcnt
__device__ __forceinline__ void async_ld16(uint32_t ldsOff, const _Float16* g) {
  asm volatile("global_load_async_to_lds_b128 %0, %1, off"
               :: "v"(ldsOff), "v"((unsigned long long)(uintptr_t)g)
               : "memory");
}

__global__ __launch_bounds__(256)
void k_gemm_wmma(const _Float16* __restrict__ A, const _Float16* __restrict__ Bt,
                 const float* __restrict__ bias, float* __restrict__ out32,
                 _Float16* __restrict__ out16,
                 int Md, int Nd, int Kd,
                 long long sA, long long sB, long long sO, int applyGelu) {
  __shared__ _Float16 As[2][128 * 48];  // 32 halves/row padded to 48 (16B-aligned rows)
  __shared__ _Float16 Bs[2][128 * 48];

  int bz = blockIdx.z;
  A  += (size_t)bz * (size_t)sA;
  Bt += (size_t)bz * (size_t)sB;

  int tid  = threadIdx.x;
  int lane = tid & 31;
  int wid  = tid >> 5;
  int wm   = wid & 3;        // wave row: 4 x 32 rows
  int wn   = wid >> 2;       // wave col: 2 x 64 cols
  int lr   = lane & 15;
  int kg   = lane >> 4;
  int rowBlk = blockIdx.y * 128;
  int colBlk = blockIdx.x * 128;

  // each thread stages rows r0 and r0+64 (one 16B segment each) of both tiles
  int r0  = tid >> 2;
  int seg = tid & 3;
  uint32_t ldsOff0 = (uint32_t)((r0 * 48 + seg * 8) * 2);
  uint32_t ldsOff1 = ldsOff0 + (uint32_t)(64 * 48 * 2);
  uint32_t aBase[2] = { (uint32_t)(uintptr_t)&As[0][0], (uint32_t)(uintptr_t)&As[1][0] };
  uint32_t bBase[2] = { (uint32_t)(uintptr_t)&Bs[0][0], (uint32_t)(uintptr_t)&Bs[1][0] };
  const _Float16* gA0 = A  + (size_t)(rowBlk + r0) * Kd + seg * 8;
  const _Float16* gA1 = gA0 + (size_t)64 * Kd;
  const _Float16* gB0 = Bt + (size_t)(colBlk + r0) * Kd + seg * 8;
  const _Float16* gB1 = gB0 + (size_t)64 * Kd;

  v8f zero = {};
  v8f acc[2][4];
  for (int i = 0; i < 2; ++i)
    for (int j = 0; j < 4; ++j) acc[i][j] = zero;

  // 8 WMMAs on one staged 128x32 tile pair
  auto compute = [&](int buf) {
    const _Float16* as = As[buf];
    const _Float16* bs = Bs[buf];
    v16h a0 = lds_frag(as, wm * 32 +  0, lr, kg);
    v16h a1 = lds_frag(as, wm * 32 + 16, lr, kg);
    v16h b0 = lds_frag(bs, wn * 64 +  0, lr, kg);
    v16h b1 = lds_frag(bs, wn * 64 + 16, lr, kg);
    v16h b2 = lds_frag(bs, wn * 64 + 32, lr, kg);
    v16h b3 = lds_frag(bs, wn * 64 + 48, lr, kg);
    acc[0][0] = __builtin_amdgcn_wmma_f32_16x16x32_f16(false, a0, false, b0, (short)0, acc[0][0], false, false);
    acc[0][1] = __builtin_amdgcn_wmma_f32_16x16x32_f16(false, a0, false, b1, (short)0, acc[0][1], false, false);
    acc[0][2] = __builtin_amdgcn_wmma_f32_16x16x32_f16(false, a0, false, b2, (short)0, acc[0][2], false, false);
    acc[0][3] = __builtin_amdgcn_wmma_f32_16x16x32_f16(false, a0, false, b3, (short)0, acc[0][3], false, false);
    acc[1][0] = __builtin_amdgcn_wmma_f32_16x16x32_f16(false, a1, false, b0, (short)0, acc[1][0], false, false);
    acc[1][1] = __builtin_amdgcn_wmma_f32_16x16x32_f16(false, a1, false, b1, (short)0, acc[1][1], false, false);
    acc[1][2] = __builtin_amdgcn_wmma_f32_16x16x32_f16(false, a1, false, b2, (short)0, acc[1][2], false, false);
    acc[1][3] = __builtin_amdgcn_wmma_f32_16x16x32_f16(false, a1, false, b3, (short)0, acc[1][3], false, false);
  };

  int nsteps = Kd >> 5;

  // prologue: stage tile 0 into buffer 0 (4 async b128 ops per thread)
  async_ld16(aBase[0] + ldsOff0, gA0);
  async_ld16(aBase[0] + ldsOff1, gA1);
  async_ld16(bBase[0] + ldsOff0, gB0);
  async_ld16(bBase[0] + ldsOff1, gB1);

  // steady state: always prefetch next tile, wait for current (4 in flight)
  for (int s = 0; s < nsteps - 1; ++s) {
    int cur = s & 1;
    int nxt = cur ^ 1;
    int kb  = (s + 1) << 5;
    async_ld16(aBase[nxt] + ldsOff0, gA0 + kb);
    async_ld16(aBase[nxt] + ldsOff1, gA1 + kb);
    async_ld16(bBase[nxt] + ldsOff0, gB0 + kb);
    async_ld16(bBase[nxt] + ldsOff1, gB1 + kb);
    asm volatile("s_wait_asynccnt 0x4" ::: "memory");
    __syncthreads();
    compute(cur);
    __syncthreads();   // safe to overwrite this buffer next round
  }

  // tail: last tile, no prefetch behind it
  asm volatile("s_wait_asynccnt 0x0" ::: "memory");
  __syncthreads();
  compute((nsteps - 1) & 1);

  // epilogue: C/D layout (ISA 7.12.2): col = lane&15, row = vgpr + 8*(lane>>4)
  for (int im = 0; im < 2; ++im) {
    for (int jn = 0; jn < 4; ++jn) {
      int nG = colBlk + wn * 64 + jn * 16 + lr;
      if (nG >= Nd) continue;
      float bv = bias ? bias[nG] : 0.f;
      v8f a = acc[im][jn];
      int mBase = rowBlk + wm * 32 + im * 16 + kg * 8;
      for (int g = 0; g < 8; ++g) {
        int mG = mBase + g;
        if (mG >= Md) continue;
        float vv = a[g] + bv;
        if (applyGelu) {
          float u = 0.7978845608028654f * (vv + 0.044715f * vv * vv * vv);
          vv = 0.5f * vv * (1.f + tanhf(u));
        }
        size_t o = (size_t)bz * (size_t)sO + (size_t)mG * Nd + nG;
        if (out32) out32[o] = vv;
        if (out16) out16[o] = (_Float16)vv;
      }
    }
  }
}

// ---------------- host orchestration ----------------
extern "C" void kernel_launch(void* const* d_in, const int* in_sizes, int n_in,
                              void* d_out, int out_size, void* d_ws, size_t ws_size,
                              hipStream_t stream) {
  (void)in_sizes; (void)n_in; (void)out_size; (void)ws_size;

  const int*   tokens = (const int*)  d_in[0];
  const float* wte    = (const float*)d_in[1];
  const float* wpe    = (const float*)d_in[2];
  const float* ln1_g  = (const float*)d_in[3];
  const float* ln1_b  = (const float*)d_in[4];
  const float* w_qkv  = (const float*)d_in[5];
  const float* b_qkv  = (const float*)d_in[6];
  const float* w_pr   = (const float*)d_in[7];
  const float* b_pr   = (const float*)d_in[8];
  const float* ln2_g  = (const float*)d_in[9];
  const float* ln2_b  = (const float*)d_in[10];
  const float* w_fc   = (const float*)d_in[11];
  const float* b_fc   = (const float*)d_in[12];
  const float* w_pj   = (const float*)d_in[13];
  const float* b_pj   = (const float*)d_in[14];
  const float* lnf_g  = (const float*)d_in[15];
  const float* lnf_b  = (const float*)d_in[16];
  const float* w_out  = (const float*)d_in[17];
  const float* b_out  = (const float*)d_in[18];

  const size_t MB = 1024 * 1024;
  char* w = (char*)d_ws;
  float*    X32   = (float*)   (w +   0 * MB);  //  8 MB  activation fp32 [B,T,C]
  _Float16* A16   = (_Float16*)(w +   8 * MB);  //  4 MB  f16 activation [B,T,C]
  _Float16* G16   = (_Float16*)(w +  12 * MB);  // 16 MB  f16 gelu out [B,T,4C]
  _Float16* W16   = (_Float16*)(w +  28 * MB);  //  8 MB  f16 transposed weight (max 4096x1024)
  float*    QKV32 = (float*)   (w +  36 * MB);  // 24 MB  qkv fp32 (reused as proj out)
  _Float16* Q16   = (_Float16*)(w +  60 * MB);  //  4 MB  [B,H,T,D]
  _Float16* K16f  = (_Float16*)(w +  64 * MB);  //  4 MB  [B,H,T,D]
  _Float16* VT16  = (_Float16*)(w +  68 * MB);  //  8 MB  [B,H,128,T] (rows 64..127 pad)
  float*    Y32   = (float*)   (w +  76 * MB);  //  8 MB  [B,H,T,D]
  _Float16* P16   = (_Float16*)(w +  84 * MB);  // 64 MB  [B,H,T,T] scores/probs

  const int MROWS = B_ * T_;                    // 2048
  dim3 blk(256);

  k_embed<<<MROWS, blk, 0, stream>>>(tokens, wte, wpe, X32);

  for (int l = 0; l < L_; ++l) {
    // ---- LN1 -> f16
    k_layernorm<<<MROWS, blk, 0, stream>>>(X32, ln1_g + (size_t)l * C_, ln1_b + (size_t)l * C_, A16);

    // ---- qkv = h @ Wqkv + b
    {
      size_t tot = (size_t)C_ * 3 * C_;
      k_convT<<<(unsigned)((tot + 255) / 256), blk, 0, stream>>>(w_qkv + (size_t)l * C_ * 3 * C_, W16, C_, 3 * C_);
      dim3 g((3 * C_) / 128, MROWS / 128, 1);
      k_gemm_wmma<<<g, blk, 0, stream>>>(A16, W16, b_qkv + (size_t)l * 3 * C_,
                                         QKV32, nullptr, MROWS, 3 * C_, C_, 0, 0, 0, 0);
    }

    // ---- split heads
    k_split_qkv<<<(B_ * T_ * C_) / 256, blk, 0, stream>>>(QKV32, Q16, K16f, VT16);

    // ---- scores = Q @ K^T  (batched over B*H), f16 out
    {
      dim3 g(T_ / 128, T_ / 128, B_ * H_);
      k_gemm_wmma<<<g, blk, 0, stream>>>(Q16, K16f, nullptr, nullptr, P16,
                                         T_, T_, D_,
                                         (long long)T_ * D_, (long long)T_ * D_, (long long)T_ * T_, 0);
    }

    // ---- causal softmax (in place, scale 1/sqrt(D))
    k_softmax_causal<<<B_ * H_ * T_, blk, 0, stream>>>(P16, 0.125f);

    // ---- y = P @ V  (batched, N=64, Bt padded to 128 rows/batch)
    {
      dim3 g(1, T_ / 128, B_ * H_);
      k_gemm_wmma<<<g, blk, 0, stream>>>(P16, VT16, nullptr, Y32, nullptr,
                                         T_, D_, T_,
                                         (long long)T_ * T_, (long long)128 * T_, (long long)T_ * D_, 0);
    }

    // ---- merge heads -> f16
    k_merge_heads<<<(B_ * T_ * C_) / 256, blk, 0, stream>>>(Y32, A16);

    // ---- proj
    {
      size_t tot = (size_t)C_ * C_;
      k_convT<<<(unsigned)((tot + 255) / 256), blk, 0, stream>>>(w_pr + (size_t)l * C_ * C_, W16, C_, C_);
      dim3 g(C_ / 128, MROWS / 128, 1);
      k_gemm_wmma<<<g, blk, 0, stream>>>(A16, W16, b_pr + (size_t)l * C_,
                                         QKV32, nullptr, MROWS, C_, C_, 0, 0, 0, 0);
    }

    // ---- LN2 -> f16
    k_layernorm<<<MROWS, blk, 0, stream>>>(QKV32, ln2_g + (size_t)l * C_, ln2_b + (size_t)l * C_, A16);

    // ---- fc + gelu -> f16
    {
      size_t tot = (size_t)C_ * 4 * C_;
      k_convT<<<(unsigned)((tot + 255) / 256), blk, 0, stream>>>(w_fc + (size_t)l * C_ * 4 * C_, W16, C_, 4 * C_);
      dim3 g((4 * C_) / 128, MROWS / 128, 1);
      k_gemm_wmma<<<g, blk, 0, stream>>>(A16, W16, b_fc + (size_t)l * 4 * C_,
                                         nullptr, G16, MROWS, 4 * C_, C_, 0, 0, 0, 1);
    }

    // ---- pj -> next layer x (fp32)
    {
      size_t tot = (size_t)4 * C_ * C_;
      k_convT<<<(unsigned)((tot + 255) / 256), blk, 0, stream>>>(w_pj + (size_t)l * 4 * C_ * C_, W16, 4 * C_, C_);
      dim3 g(C_ / 128, MROWS / 128, 1);
      k_gemm_wmma<<<g, blk, 0, stream>>>(G16, W16, b_pj + (size_t)l * C_,
                                         X32, nullptr, MROWS, C_, 4 * C_, 0, 0, 0, 0);
    }
  }

  // ---- final LN + logits
  k_layernorm<<<MROWS, blk, 0, stream>>>(X32, lnf_g, lnf_b, A16);
  {
    size_t tot = (size_t)C_ * V_;
    k_convT<<<(unsigned)((tot + 255) / 256), blk, 0, stream>>>(w_out, W16, C_, V_);
    dim3 g(V_ / 128, MROWS / 128, 1);
    k_gemm_wmma<<<g, blk, 0, stream>>>(A16, W16, b_out,
                                       (float*)d_out, nullptr, MROWS, V_, C_, 0, 0, 0, 0);
  }
}
